// BiDynamicRNNLayer_20495583937037
// MI455X (gfx1250) — matrix operand order repaired and checked by reading, not compiled, not run on hardware
//
#include <hip/hip_runtime.h>
#include <hip/hip_bf16.h>
#include <cstdint>

// ---------------------------------------------------------------------------
// BiDynamicRNN (bidirectional LSTM), B=32, T=512, F=H=512, out [B,T,2H] f32.
// bf16 WMMA (v_wmma_f32_16x16x32_bf16) for per-step gate GEMM
// [32,1024]@[1024,2048]; weights pre-packed into WMMA B-fragment layout;
// x pre-converted (and seq-reversed for bw) to time-major bf16; per-step
// kernel fuses GEMM + LDS gate exchange + activations + state update.
// K-loop split (x half / h half) to kill address selects; dual accumulators
// to break the WMMA->WMMA dependency chain.
// ---------------------------------------------------------------------------

#define BB 32
#define TT 512
#define FF 512
#define HH 512
#define KK (FF + HH)       // 1024
#define NN (4 * HH)        // 2048
#define NKB (KK / 32)      // 32 k-steps of 32

typedef __attribute__((ext_vector_type(16))) __bf16 v16bf;
typedef __attribute__((ext_vector_type(8)))  __bf16 v8bf;
typedef __attribute__((ext_vector_type(8)))  float  v8f;

union AFrag { v16bf v; v8bf h[2]; };

__device__ __forceinline__ float sigmoidf_(float x) {
    return 1.0f / (1.0f + __expf(-x));
}

// ------------------------- setup: pack weights -----------------------------
// Bpack[dir][ntile][kb][lane][e] : lane holds column n = ntile*16 + (lane&15),
// k = kb*32 + (e<8 ? 8*(lane>>4)+e : 16+8*(lane>>4)+(e-8))   (dense bf16 B 32x16 fragment layout)
__global__ void pack_weights_kernel(const float* __restrict__ Wfw,
                                    const float* __restrict__ Wbw,
                                    __bf16* __restrict__ Bpack) {
    size_t p = (size_t)blockIdx.x * blockDim.x + threadIdx.x;  // < 2*1024*2048
    int e     = (int)(p & 15);
    int lane  = (int)((p >> 4) & 31);
    int kb    = (int)((p >> 9) & 31);
    int ntile = (int)((p >> 14) & 127);
    int dir   = (int)(p >> 21);
    int k = kb * 32 + ((e < 8) ? ((lane >> 4) * 8 + e) : (16 + (lane >> 4) * 8 + (e - 8)));
    int n = ntile * 16 + (lane & 15);
    const float* W = dir ? Wbw : Wfw;
    Bpack[p] = (__bf16)W[(size_t)k * NN + n];
}

// --------------------- setup: convert x (+ bw reversal) --------------------
// xbf[dir][t][b][f] bf16, time-major. bw uses reverse_seq gather per batch.
__global__ void convert_x_kernel(const float* __restrict__ x,
                                 const int* __restrict__ seqlen,
                                 __bf16* __restrict__ xbf) {
    size_t p = (size_t)blockIdx.x * blockDim.x + threadIdx.x;  // < 2*512*32*512
    int f   = (int)(p & 511);
    int b   = (int)((p >> 9) & 31);
    int t   = (int)((p >> 14) & 511);
    int dir = (int)(p >> 23);
    int ts = t;
    if (dir) {
        int len = seqlen[b];
        ts = (t < len) ? (len - 1 - t) : t;
    }
    xbf[p] = (__bf16)x[((size_t)b * TT + ts) * FF + f];
}

// --------------------------- setup: zero state -----------------------------
__global__ void zero_state_kernel(__bf16* __restrict__ hbufs,
                                  float* __restrict__ cbuf) {
    int tid = blockIdx.x * blockDim.x + threadIdx.x;  // 65536 threads
    if (tid < 2 * 2 * BB * HH) hbufs[tid] = (__bf16)0.0f;
    if (tid < 2 * BB * HH)     cbuf[tid]  = 0.0f;
}

// ------------------------------ LSTM step ----------------------------------
// grid (32 h-tiles, 2 directions), block 256 = 8 waves.
// wave -> (mtile = w&1, gate = w>>1); each wave: one 16x16 f32 tile over K=1024.
__global__ void lstm_step_kernel(const __bf16* __restrict__ xbf,
                                 const __bf16* __restrict__ Bpack,
                                 const float* __restrict__ bias_fw,
                                 const float* __restrict__ bias_bw,
                                 const int* __restrict__ seqlen,
                                 __bf16* __restrict__ hbufs,
                                 float* __restrict__ cbuf,
                                 float* __restrict__ out,
                                 int s) {
    const int dir   = blockIdx.y;
    const int ht    = blockIdx.x;          // hidden-column tile: columns h0..h0+15
    const int h0    = ht * 16;
    const int wave  = threadIdx.x >> 5;
    const int lane  = threadIdx.x & 31;
    const int mtile = wave & 1;            // batch rows [mtile*16, +16)
    const int gate  = wave >> 1;           // 0:i 1:j 2:f 3:o
    const int ntile = gate * 32 + ht;      // column tile in [0,128)

    const int pbuf = s & 1;
    const __bf16* __restrict__ xrow_base =
        xbf + (((size_t)dir * TT + s) * BB) * FF;
    const __bf16* __restrict__ hprev =
        hbufs + (size_t)(dir * 2 + pbuf) * BB * HH;
    __bf16* __restrict__ hnext =
        hbufs + (size_t)(dir * 2 + (pbuf ^ 1)) * BB * HH;

    // ---- GEMM: acc[16x16] += A[16x1024] * B[1024x16] via 32 WMMA k-steps ---
    const int m = mtile * 16 + (lane & 15);
    const int off = (lane < 16) ? 0 : 8;
    const __bf16* __restrict__ xrow = xrow_base + (size_t)m * FF + off;
    const __bf16* __restrict__ hrow = hprev + (size_t)m * HH + off;
    const __bf16* __restrict__ bfrag =
        Bpack + (size_t)dir * KK * NN + (((size_t)ntile * NKB) * 32 + lane) * 16;

    v8f acc0 = {};
    v8f acc1 = {};
    // k-steps 0..15: A rows come from x_t   (k0 = kb*32 in [0,512))
#pragma unroll
    for (int kb = 0; kb < 16; ++kb) {
        AFrag a;
        a.h[0] = *(const v8bf*)(xrow + kb * 32);
        a.h[1] = *(const v8bf*)(xrow + kb * 32 + 16);
        v16bf bv = *(const v16bf*)(bfrag + (size_t)kb * 512);
        if (kb & 1)
            acc1 = __builtin_amdgcn_wmma_f32_16x16x32_bf16(
                false, a.v, false, bv, (short)0, acc1, false, false);
        else
            acc0 = __builtin_amdgcn_wmma_f32_16x16x32_bf16(
                false, a.v, false, bv, (short)0, acc0, false, false);
    }
    // k-steps 16..31: A rows come from h_prev (k0-512 in [0,512))
#pragma unroll
    for (int kb = 0; kb < 16; ++kb) {
        AFrag a;
        a.h[0] = *(const v8bf*)(hrow + kb * 32);
        a.h[1] = *(const v8bf*)(hrow + kb * 32 + 16);
        v16bf bv = *(const v16bf*)(bfrag + (size_t)(kb + 16) * 512);
        if (kb & 1)
            acc1 = __builtin_amdgcn_wmma_f32_16x16x32_bf16(
                false, a.v, false, bv, (short)0, acc1, false, false);
        else
            acc0 = __builtin_amdgcn_wmma_f32_16x16x32_bf16(
                false, a.v, false, bv, (short)0, acc0, false, false);
    }

    // ---- exchange gate tiles through LDS ----
    __shared__ float g_lds[4][BB][16];
    const int nloc = lane & 15;
#pragma unroll
    for (int r = 0; r < 8; ++r) {
        int mloc = mtile * 16 + ((lane < 16) ? r : 8 + r);
        g_lds[gate][mloc][nloc] = acc0[r] + acc1[r];
    }
    __syncthreads();

    // ---- activations + state update: 512 (b,n) elems over 256 threads -----
    const float* __restrict__ bias = dir ? bias_bw : bias_fw;
#pragma unroll
    for (int q = 0; q < 2; ++q) {
        const int idx = threadIdx.x + q * 256;  // 0..511
        const int b = idx >> 4;
        const int n = idx & 15;
        const float gi = g_lds[0][b][n] + bias[0 * HH + h0 + n];
        const float gj = g_lds[1][b][n] + bias[1 * HH + h0 + n];
        const float gf = g_lds[2][b][n] + bias[2 * HH + h0 + n];
        const float go = g_lds[3][b][n] + bias[3 * HH + h0 + n];
        const int len = seqlen[b];
        const bool msk = (s < len);

        const size_t cidx = ((size_t)dir * BB + b) * HH + h0 + n;
        const float c_old = cbuf[cidx];
        const float c_new = sigmoidf_(gf + 1.0f) * c_old
                          + sigmoidf_(gi) * tanhf(gj);
        const float h_new = sigmoidf_(go) * tanhf(c_new);
        const float h_old = (float)hprev[(size_t)b * HH + h0 + n];

        cbuf[cidx] = msk ? c_new : c_old;
        hnext[(size_t)b * HH + h0 + n] = (__bf16)(msk ? h_new : h_old);

        // output scatter (bw applies reverse_seq involution)
        const int tpos = (dir == 0) ? s : (msk ? (len - 1 - s) : s);
        out[((size_t)b * TT + tpos) * (2 * HH) + dir * HH + h0 + n] =
            msk ? h_new : 0.0f;
    }
}

// ---------------------------------------------------------------------------
extern "C" void kernel_launch(void* const* d_in, const int* in_sizes, int n_in,
                              void* d_out, int out_size, void* d_ws, size_t ws_size,
                              hipStream_t stream) {
    const float* x       = (const float*)d_in[0];   // [B,T,F]
    const int*   seqlen  = (const int*)d_in[1];     // [B]
    const float* W_fw    = (const float*)d_in[2];   // [1024,2048]
    const float* b_fw    = (const float*)d_in[3];   // [2048]
    const float* W_bw    = (const float*)d_in[4];
    const float* b_bw    = (const float*)d_in[5];
    float*       out     = (float*)d_out;           // [B,T,2H]

    char* ws = (char*)d_ws;
    const size_t sz_Bpack = (size_t)2 * KK * NN * sizeof(__bf16);        // 8 MB
    const size_t sz_xbf   = (size_t)2 * TT * BB * FF * sizeof(__bf16);   // 32 MB
    const size_t sz_h     = (size_t)2 * 2 * BB * HH * sizeof(__bf16);    // 128 KB
    __bf16* Bpack = (__bf16*)(ws);
    __bf16* xbf   = (__bf16*)(ws + sz_Bpack);
    __bf16* hbufs = (__bf16*)(ws + sz_Bpack + sz_xbf);
    float*  cbuf  = (float*) (ws + sz_Bpack + sz_xbf + sz_h);

    // setup
    {
        size_t n = (size_t)2 * KK * NN;                  // 4,194,304
        pack_weights_kernel<<<(int)(n / 256), 256, 0, stream>>>(W_fw, W_bw, Bpack);
    }
    {
        size_t n = (size_t)2 * TT * BB * FF;             // 16,777,216
        convert_x_kernel<<<(int)(n / 256), 256, 0, stream>>>(x, seqlen, xbf);
    }
    zero_state_kernel<<<256, 256, 0, stream>>>(hbufs, cbuf);

    // sequential recurrence: both directions fused per step launch
    dim3 grid(HH / 16, 2);  // 32 h-tiles x 2 directions
    for (int s = 0; s < TT; ++s) {
        lstm_step_kernel<<<grid, 256, 0, stream>>>(
            xbf, Bpack, b_fw, b_bw, seqlen, hbufs, cbuf, out, s);
    }

    (void)in_sizes; (void)n_in; (void)out_size; (void)ws_size;
}
